// TripletLoss_2937757630817
// MI455X (gfx1250) — compile-verified, matrix operand order checked by reading, not run on hardware
//
#include <hip/hip_runtime.h>
#include <hip/hip_bf16.h>

// ---------------------------------------------------------------------------
// Triplet loss vs gallery bank, fused GEMM + masked min/max reduction.
//   sim = inputs[1024,512] @ features[32768,512]^T
//   pos = min over same-label cols (excluding own idx), neg = max over others
//   out = mean(relu(neg - pos + 0.3))
//
// Fast path (needs ~66MB workspace):
//   1) tl_convert: split f32 -> bf16 hi + bf16 lo once, into workspace.
//   2) tl_main_async: 64x64 block GEMM. Tiles stream global->LDS with
//      GLOBAL_LOAD_ASYNC_TO_LDS_B128 (ASYNCcnt), double-buffered, one
//      barrier/stage. bf16x3 WMMA (hi*hi + hi*lo + lo*hi ~ f32 accuracy),
//      fused masked min/max epilogue via order-preserving u32 atomics.
// Fallback (small workspace): fused-conversion kernel.
// ---------------------------------------------------------------------------

#define N_BATCH   1024
#define N_GALLERY 32768
#define D_EMB     512
#define MARGIN    0.3f

#define K_STEP    32      // K per WMMA stage
// fast-path tile
#define PB_M      64
#define PB_N      64
// fallback tile
#define FB_M      32
#define FB_N      64

typedef __attribute__((ext_vector_type(16))) __bf16 v16bf;
typedef __attribute__((ext_vector_type(4)))  __bf16 v4bf;
typedef __attribute__((ext_vector_type(8)))  float  v8f;

// Order-preserving float -> uint mapping: enc(a) < enc(b)  <=>  a < b.
__device__ __forceinline__ unsigned enc_f32(float f) {
    unsigned u = __float_as_uint(f);
    return (u & 0x80000000u) ? ~u : (u | 0x80000000u);
}
__device__ __forceinline__ float dec_f32(unsigned u) {
    return (u & 0x80000000u) ? __uint_as_float(u & 0x7fffffffu)
                             : __uint_as_float(~u);
}

#define ENC_POS_INF 0xFF800000u   // enc(+inf)  (identity for min)
#define ENC_NEG_INF 0x007FFFFFu   // enc(-inf)  (identity for max)

// ---- CDNA5 async global->LDS DMA (ASYNCcnt-tracked), via inline asm --------
// vdst = VGPR holding LDS byte offset, vaddr = 64-bit global address.
__device__ __forceinline__ void async_copy_b128(unsigned lds_off, const void* gp) {
    asm volatile("global_load_async_to_lds_b128 %0, %1, off"
                 :: "v"(lds_off), "v"((unsigned long long)(size_t)gp)
                 : "memory");
}
#define WAIT_ASYNCCNT0() asm volatile("s_wait_asynccnt 0x0" ::: "memory")

// Low 32 bits of a generic pointer to __shared__ == LDS byte offset.
__device__ __forceinline__ unsigned lds_off32(const void* p) {
    return (unsigned)(size_t)p;
}

__global__ void tl_init(unsigned* __restrict__ gpos, unsigned* __restrict__ gneg) {
    int i = blockIdx.x * blockDim.x + threadIdx.x;
    if (i < N_BATCH) { gpos[i] = ENC_POS_INF; gneg[i] = ENC_NEG_INF; }
}

// ---------------- pre-split pass: f32 -> bf16 hi/lo (one float4/thread) -----
__global__ void tl_convert(const float* __restrict__ src,
                           unsigned short* __restrict__ hi_,
                           unsigned short* __restrict__ lo_, int n4) {
    int i = blockIdx.x * blockDim.x + threadIdx.x;
    if (i >= n4) return;
    const float4 v = reinterpret_cast<const float4*>(src)[i];
    const float* vp = &v.x;
    v4bf h, l;
    #pragma unroll
    for (int j = 0; j < 4; ++j) {
        float  x  = vp[j];
        __bf16 hh = (__bf16)x;
        h[j] = hh;
        l[j] = (__bf16)(x - (float)hh);
    }
    *reinterpret_cast<v4bf*>(reinterpret_cast<__bf16*>(hi_) + 4 * (size_t)i) = h;
    *reinterpret_cast<v4bf*>(reinterpret_cast<__bf16*>(lo_) + 4 * (size_t)i) = l;
}

// ---------------- fast path: preconverted bf16, async-LDS double buffer -----
__global__ __launch_bounds__(256)
void tl_main_async(const unsigned short* __restrict__ Ahi_,
                   const unsigned short* __restrict__ Alo_,
                   const unsigned short* __restrict__ Bhi_,
                   const unsigned short* __restrict__ Blo_,
                   const int* __restrict__ tgt, const int* __restrict__ sidx,
                   const int* __restrict__ flab,
                   unsigned* __restrict__ gpos, unsigned* __restrict__ gneg)
{
    const __bf16* Ahi = reinterpret_cast<const __bf16*>(Ahi_);
    const __bf16* Alo = reinterpret_cast<const __bf16*>(Alo_);
    const __bf16* Bhi = reinterpret_cast<const __bf16*>(Bhi_);
    const __bf16* Blo = reinterpret_cast<const __bf16*>(Blo_);

    __shared__ __align__(32) __bf16 sAhi[2][PB_M * K_STEP];   // 4KB per buf
    __shared__ __align__(32) __bf16 sAlo[2][PB_M * K_STEP];
    __shared__ __align__(32) __bf16 sBhi[2][PB_N * K_STEP];
    __shared__ __align__(32) __bf16 sBlo[2][PB_N * K_STEP];
    __shared__ unsigned lpos[PB_M];
    __shared__ unsigned lneg[PB_M];

    const int tid   = threadIdx.x;
    const int nbase = blockIdx.x * PB_N;
    const int mbase = blockIdx.y * PB_M;

    if (tid < PB_M) { lpos[tid] = ENC_POS_INF; lneg[tid] = ENC_NEG_INF; }

    // copy mapping: 256 threads move 64 rows x 32 cols bf16 (one b128 each)
    const int crow = tid >> 2;          // 0..63
    const int c8   = (tid & 3) << 3;    // 0,8,16,24

    // wave -> two 16x16 C tiles (M pair) x one N tile
    const int wave = tid >> 5;
    const int lane = tid & 31;
    const int ni   = wave & 3;          // 0..3
    const int mg   = wave >> 2;         // 0..1 -> M tiles 2*mg, 2*mg+1
    const int lm   = lane & 15;
    const int lh   = lane >> 4;

    // per-thread global source rows (advance by k along the row)
    const __bf16* gAhi = Ahi + (size_t)(mbase + crow) * D_EMB + c8;
    const __bf16* gAlo = Alo + (size_t)(mbase + crow) * D_EMB + c8;
    const __bf16* gBhi = Bhi + (size_t)(nbase + crow) * D_EMB + c8;
    const __bf16* gBlo = Blo + (size_t)(nbase + crow) * D_EMB + c8;

    // per-thread LDS destinations (buffer 0); buffer 1 is +BUF_BYTES
    const unsigned BUF_BYTES = PB_M * K_STEP * 2;   // 4096 (same for A and B)
    const unsigned dAhi = lds_off32(&sAhi[0][crow * K_STEP + c8]);
    const unsigned dAlo = lds_off32(&sAlo[0][crow * K_STEP + c8]);
    const unsigned dBhi = lds_off32(&sBhi[0][crow * K_STEP + c8]);
    const unsigned dBlo = lds_off32(&sBlo[0][crow * K_STEP + c8]);

    // prologue: stage 0 -> buffer 0
    async_copy_b128(dAhi, gAhi);
    async_copy_b128(dAlo, gAlo);
    async_copy_b128(dBhi, gBhi);
    async_copy_b128(dBlo, gBlo);
    WAIT_ASYNCCNT0();
    __syncthreads();

    v8f c0 = {0.f, 0.f, 0.f, 0.f, 0.f, 0.f, 0.f, 0.f};
    v8f c1 = {0.f, 0.f, 0.f, 0.f, 0.f, 0.f, 0.f, 0.f};

    const int fr0 = ((mg * 32)      + lm) * 2 + lh;   // A tile 2*mg   (v16bf idx)
    const int fr1 = ((mg * 32 + 16) + lm) * 2 + lh;   // A tile 2*mg+1
    const int frb = (ni * 16 + lm) * 2 + lh;          // B tile ni

    #pragma unroll 2
    for (int ks = 0; ks < D_EMB / K_STEP; ++ks) {
        const int cur = ks & 1;
        const int nxt = cur ^ 1;
        const int k1  = (ks + 1) * K_STEP;
        const bool has_next = (k1 < D_EMB);

        // kick off next stage's DMA into the other buffer (no VGPR staging):
        // previous iteration's barrier guarantees nobody still reads `nxt`.
        if (has_next) {
            const unsigned bo = (unsigned)nxt * BUF_BYTES;
            async_copy_b128(dAhi + bo, gAhi + k1);
            async_copy_b128(dAlo + bo, gAlo + k1);
            async_copy_b128(dBhi + bo, gBhi + k1);
            async_copy_b128(dBlo + bo, gBlo + k1);
        }

        // compute from current buffer while the DMA runs
        const v16bf* pAh = reinterpret_cast<const v16bf*>(sAhi[cur]);
        const v16bf* pAl = reinterpret_cast<const v16bf*>(sAlo[cur]);
        const v16bf* pBh = reinterpret_cast<const v16bf*>(sBhi[cur]);
        const v16bf* pBl = reinterpret_cast<const v16bf*>(sBlo[cur]);
        const v16bf a0h = pAh[fr0], a0l = pAl[fr0];
        const v16bf a1h = pAh[fr1], a1l = pAl[fr1];
        const v16bf bh  = pBh[frb], bl  = pBl[frb];

        c0 = __builtin_amdgcn_wmma_f32_16x16x32_bf16(false, a0h, false, bh,
                                                     (short)0, c0, false, false);
        c0 = __builtin_amdgcn_wmma_f32_16x16x32_bf16(false, a0h, false, bl,
                                                     (short)0, c0, false, false);
        c0 = __builtin_amdgcn_wmma_f32_16x16x32_bf16(false, a0l, false, bh,
                                                     (short)0, c0, false, false);
        c1 = __builtin_amdgcn_wmma_f32_16x16x32_bf16(false, a1h, false, bh,
                                                     (short)0, c1, false, false);
        c1 = __builtin_amdgcn_wmma_f32_16x16x32_bf16(false, a1h, false, bl,
                                                     (short)0, c1, false, false);
        c1 = __builtin_amdgcn_wmma_f32_16x16x32_bf16(false, a1l, false, bh,
                                                     (short)0, c1, false, false);

        // own DMAs done -> barrier makes the whole block's stage k+1 visible
        if (has_next) WAIT_ASYNCCNT0();
        __syncthreads();
    }

    // ---- epilogue: mask + per-row min/max reduction ----
    // C layout: element r holds M = r + 8*lh, N = lm within the 16x16 tile.
    const int gcol = nbase + ni * 16 + lm;
    const int fl   = flab[gcol];

    #pragma unroll
    for (int t = 0; t < 2; ++t) {
        const v8f cc = t ? c1 : c0;
        const int mt = mg * 2 + t;
        #pragma unroll
        for (int r = 0; r < 8; ++r) {
            const int lrow = mt * 16 + lh * 8 + r;     // 0..63
            const int grow = mbase + lrow;
            const int tg   = tgt[grow];
            const int si   = sidx[grow];
            const float s  = cc[r];
            float pv = (tg == fl && si != gcol) ? s :  __builtin_huge_valf();
            float nv = (tg != fl)               ? s : -__builtin_huge_valf();
            #pragma unroll
            for (int off = 8; off >= 1; off >>= 1) {
                pv = fminf(pv, __shfl_xor(pv, off, 32));
                nv = fmaxf(nv, __shfl_xor(nv, off, 32));
            }
            if (lm == 0) {
                atomicMin(&lpos[lrow], enc_f32(pv));
                atomicMax(&lneg[lrow], enc_f32(nv));
            }
        }
    }

    __syncthreads();
    if (tid < PB_M) {
        atomicMin(&gpos[mbase + tid], lpos[tid]);
        atomicMax(&gneg[mbase + tid], lneg[tid]);
    }
}

// ---------------- fallback: fused conversion GEMM ---------------------------
__global__ __launch_bounds__(256)
void tl_main_fused(const float* __restrict__ inp, const float* __restrict__ feat,
                   const int* __restrict__ tgt, const int* __restrict__ sidx,
                   const int* __restrict__ flab,
                   unsigned* __restrict__ gpos, unsigned* __restrict__ gneg)
{
    __shared__ __align__(32) __bf16 sAhi[FB_M * K_STEP];
    __shared__ __align__(32) __bf16 sAlo[FB_M * K_STEP];
    __shared__ __align__(32) __bf16 sBhi[FB_N * K_STEP];
    __shared__ __align__(32) __bf16 sBlo[FB_N * K_STEP];
    __shared__ unsigned lpos[FB_M];
    __shared__ unsigned lneg[FB_M];

    const int tid   = threadIdx.x;
    const int nbase = blockIdx.x * FB_N;
    const int mbase = blockIdx.y * FB_M;

    if (tid < FB_M) { lpos[tid] = ENC_POS_INF; lneg[tid] = ENC_NEG_INF; }

    const int wave = tid >> 5;
    const int lane = tid & 31;
    const int mi   = wave & 1;
    const int ni   = wave >> 1;
    const int lm   = lane & 15;
    const int lh   = lane >> 4;

    const v16bf* pAhi = reinterpret_cast<const v16bf*>(sAhi);
    const v16bf* pAlo = reinterpret_cast<const v16bf*>(sAlo);
    const v16bf* pBhi = reinterpret_cast<const v16bf*>(sBhi);
    const v16bf* pBlo = reinterpret_cast<const v16bf*>(sBlo);

    v8f c = {0.f, 0.f, 0.f, 0.f, 0.f, 0.f, 0.f, 0.f};

    const int arow = tid >> 3;
    const int ac4  = (tid & 7) << 2;

    for (int k0 = 0; k0 < D_EMB; k0 += K_STEP) {
        __syncthreads();
        {
            const float4 v = *reinterpret_cast<const float4*>(
                &inp[(size_t)(mbase + arow) * D_EMB + k0 + ac4]);
            const float* vp = &v.x;
            #pragma unroll
            for (int j = 0; j < 4; ++j) {
                float  x  = vp[j];
                __bf16 hi = (__bf16)x;
                sAhi[arow * K_STEP + ac4 + j] = hi;
                sAlo[arow * K_STEP + ac4 + j] = (__bf16)(x - (float)hi);
            }
        }
        #pragma unroll
        for (int h = 0; h < 2; ++h) {
            const int brow = arow + 32 * h;
            const float* gp = &feat[(size_t)(nbase + brow) * D_EMB + k0 + ac4];
            const float4 v = *reinterpret_cast<const float4*>(gp);
            const float* vp = &v.x;
            #pragma unroll
            for (int j = 0; j < 4; ++j) {
                float  x  = vp[j];
                __bf16 hi = (__bf16)x;
                sBhi[brow * K_STEP + ac4 + j] = hi;
                sBlo[brow * K_STEP + ac4 + j] = (__bf16)(x - (float)hi);
            }
        }
        __syncthreads();

        const v16bf a_hi = pAhi[(mi * 16 + lm) * 2 + lh];
        const v16bf a_lo = pAlo[(mi * 16 + lm) * 2 + lh];
        const v16bf b_hi = pBhi[(ni * 16 + lm) * 2 + lh];
        const v16bf b_lo = pBlo[(ni * 16 + lm) * 2 + lh];

        c = __builtin_amdgcn_wmma_f32_16x16x32_bf16(false, a_hi, false, b_hi,
                                                    (short)0, c, false, false);
        c = __builtin_amdgcn_wmma_f32_16x16x32_bf16(false, a_hi, false, b_lo,
                                                    (short)0, c, false, false);
        c = __builtin_amdgcn_wmma_f32_16x16x32_bf16(false, a_lo, false, b_hi,
                                                    (short)0, c, false, false);
    }

    const int gcol = nbase + ni * 16 + lm;
    const int fl   = flab[gcol];
    #pragma unroll
    for (int r = 0; r < 8; ++r) {
        const int grow = mbase + mi * 16 + lh * 8 + r;
        const int tg   = tgt[grow];
        const int si   = sidx[grow];
        const float s  = c[r];
        float pv = (tg == fl && si != gcol) ? s :  __builtin_huge_valf();
        float nv = (tg != fl)               ? s : -__builtin_huge_valf();
        #pragma unroll
        for (int off = 8; off >= 1; off >>= 1) {
            pv = fminf(pv, __shfl_xor(pv, off, 32));
            nv = fmaxf(nv, __shfl_xor(nv, off, 32));
        }
        if (lm == 0) {
            atomicMin(&lpos[mi * 16 + lh * 8 + r], enc_f32(pv));
            atomicMax(&lneg[mi * 16 + lh * 8 + r], enc_f32(nv));
        }
    }

    __syncthreads();
    if (tid < FB_M) {
        atomicMin(&gpos[mbase + tid], lpos[tid]);
        atomicMax(&gneg[mbase + tid], lneg[tid]);
    }
}

__global__ void tl_final(const unsigned* __restrict__ gpos,
                         const unsigned* __restrict__ gneg,
                         float* __restrict__ out)
{
    __shared__ float wsum[8];
    float s = 0.f;
    for (int i = threadIdx.x; i < N_BATCH; i += 256) {
        float p = dec_f32(gpos[i]);
        float n = dec_f32(gneg[i]);
        float l = n - p + MARGIN;     // p == +inf (no positives) -> l = -inf -> 0
        s += (l > 0.f) ? l : 0.f;
    }
    #pragma unroll
    for (int off = 16; off >= 1; off >>= 1) s += __shfl_xor(s, off, 32);
    if ((threadIdx.x & 31) == 0) wsum[threadIdx.x >> 5] = s;
    __syncthreads();
    if (threadIdx.x < 8) {
        float t = wsum[threadIdx.x];
        #pragma unroll
        for (int off = 4; off >= 1; off >>= 1) t += __shfl_xor(t, off, 32);
        if (threadIdx.x == 0) out[0] = t / (float)N_BATCH;
    }
}

extern "C" void kernel_launch(void* const* d_in, const int* in_sizes, int n_in,
                              void* d_out, int out_size, void* d_ws, size_t ws_size,
                              hipStream_t stream) {
    (void)in_sizes; (void)n_in; (void)out_size;
    const float* inp  = (const float*)d_in[0];   // [1024, 512]
    const float* feat = (const float*)d_in[1];   // [32768, 512]
    const int*   tgt  = (const int*)d_in[2];     // [1024]
    const int*   sidx = (const int*)d_in[3];     // [1024]
    const int*   flab = (const int*)d_in[4];     // [32768]
    unsigned* gpos = (unsigned*)d_ws;            // [1024] encoded min-positive
    unsigned* gneg = gpos + N_BATCH;             // [1024] encoded max-negative
    float* out = (float*)d_out;

    const size_t a_elems  = (size_t)N_BATCH * D_EMB;
    const size_t b_elems  = (size_t)N_GALLERY * D_EMB;
    const size_t conv_off = 8192;  // after gpos/gneg, 16B-aligned
    const size_t need     = conv_off + 2 * 2 * (a_elems + b_elems);

    tl_init<<<(N_BATCH + 255) / 256, 256, 0, stream>>>(gpos, gneg);

    if (ws_size >= need) {
        unsigned short* Ahi = (unsigned short*)((char*)d_ws + conv_off);
        unsigned short* Alo = Ahi + a_elems;
        unsigned short* Bhi = Alo + a_elems;
        unsigned short* Blo = Bhi + b_elems;
        tl_convert<<<(unsigned)(a_elems / 4 / 256), 256, 0, stream>>>(
            inp, Ahi, Alo, (int)(a_elems / 4));
        tl_convert<<<(unsigned)(b_elems / 4 / 256), 256, 0, stream>>>(
            feat, Bhi, Blo, (int)(b_elems / 4));
        dim3 grid(N_GALLERY / PB_N, N_BATCH / PB_M);   // 512 x 16 blocks
        tl_main_async<<<grid, 256, 0, stream>>>(Ahi, Alo, Bhi, Blo,
                                                tgt, sidx, flab, gpos, gneg);
    } else {
        dim3 grid(N_GALLERY / FB_N, N_BATCH / FB_M);   // 512 x 32 blocks
        tl_main_fused<<<grid, 256, 0, stream>>>(inp, feat, tgt, sidx, flab,
                                                gpos, gneg);
    }
    tl_final<<<1, 256, 0, stream>>>(gpos, gneg, out);
}
